// DiT_29274497090184
// MI455X (gfx1250) — matrix operand order repaired and checked by reading, not compiled, not run on hardware
//
#include <hip/hip_runtime.h>
#include <hip/hip_bf16.h>
#include <stdint.h>

// ---------- constants (match reference shapes) ----------
#define T_TOK   4096      // B*S
#define DDIM    512
#define EEXP    16
#define IDIM    2048
#define SIDIM   1024
#define MTILE   32        // rows per M tile (2 WMMA row-frags)
#define MAXTILE 272       // 8192/32 + 16 experts of padding
#define MAXSLOT 8704      // 8192 + 16*32 pad

typedef __attribute__((ext_vector_type(16))) __bf16 v16bf;
typedef __attribute__((ext_vector_type(8)))  float  v8f;

union ABFrag { v16bf v; unsigned u[8]; uint4 q[2]; };

__device__ __forceinline__ unsigned short f2bf(float f) {
    unsigned u = __float_as_uint(f);
    unsigned r = u + 0x7FFFu + ((u >> 16) & 1u);   // round-to-nearest-even
    return (unsigned short)(r >> 16);
}

#define WMMA_BF16(A, B, C) \
    __builtin_amdgcn_wmma_f32_16x16x32_bf16(false, (A).v, false, (B).v, (short)0, (C), false, false)

// A fragment (16x32 bf16) from a row-major bf16 token row.
// Lane L: M = L&15; half = L>>4; v[0..3] = K in [k0+half*8, +8); v[4..7] = +16.
__device__ __forceinline__ void load_afrag(ABFrag& a, const uint16_t* row, int k0, int lane) {
    const uint16_t* p = row + k0 + ((lane >> 4) << 3);
    a.q[0] = *(const uint4*)(p);
    a.q[1] = *(const uint4*)(p + 16);
}
// B fragment (32x16 bf16) from TRANSPOSED bf16 weights Wt[N][K]:
// Lane L: N-row = n0+(L&15); v[0..3] = K in [k0+(L>>4)*8, +8); v[4..7] = +16.
__device__ __forceinline__ void load_bfragT(ABFrag& b, const uint16_t* nrow, int k0, int lane) {
    const uint16_t* p = nrow + k0 + ((lane >> 4) << 3);
    b.q[0] = *(const uint4*)(p);
    b.q[1] = *(const uint4*)(p + 16);
}

__device__ __forceinline__ float silu_glu(float g, float u) {
    return g * __builtin_amdgcn_rcpf(1.f + __expf(-g)) * u;
}

// ---------- kernel: zero counts ----------
__global__ void k_init(int* counts) {
    if (threadIdx.x < EEXP) counts[threadIdx.x] = 0;
}

// ---------- kernel: x (f32) -> bf16 ----------
__global__ void k_x2bf(const float* __restrict__ x, uint16_t* __restrict__ xf) {
    int i = blockIdx.x * blockDim.x + threadIdx.x;
    if (i < T_TOK * DDIM) xf[i] = f2bf(x[i]);
}

// ---------- kernel: transpose + convert W[K][N] f32 -> Wt[N][K] bf16 ----------
// grid: (N/32, K/32, numMatrices), block: (32, 8)
__global__ void k_tcvt(const float* __restrict__ W, uint16_t* __restrict__ Wt, int K, int N) {
    __shared__ uint16_t tile[32][33];
    size_t moff = (size_t)blockIdx.z * K * N;
    const float* Wp = W + moff;
    uint16_t* Wtp = Wt + moff;
    int n0 = blockIdx.x * 32, k0 = blockIdx.y * 32;
    int tx = threadIdx.x, ty = threadIdx.y;
#pragma unroll
    for (int i = 0; i < 32; i += 8)
        tile[ty + i][tx] = f2bf(Wp[(size_t)(k0 + ty + i) * N + n0 + tx]);
    __syncthreads();
#pragma unroll
    for (int i = 0; i < 32; i += 8)
        Wtp[(size_t)(n0 + ty + i) * K + k0 + tx] = tile[tx][ty + i];
}

// ---------- kernel: gating (softmax + top2 + counts) ----------
__global__ void k_gate(const float* __restrict__ x, const float* __restrict__ gw,
                       int* __restrict__ topk_idx, float* __restrict__ topk_w,
                       int* __restrict__ counts) {
    int token = blockIdx.x * 4 + (threadIdx.x >> 5);
    int lane  = threadIdx.x & 31;
    float acc[EEXP];
#pragma unroll
    for (int e = 0; e < EEXP; ++e) acc[e] = 0.f;
    const float* xr = x + (size_t)token * DDIM;
    for (int d = lane; d < DDIM; d += 32) {
        float xv = xr[d];
#pragma unroll
        for (int e = 0; e < EEXP; ++e) acc[e] += xv * gw[e * DDIM + d];
    }
#pragma unroll
    for (int e = 0; e < EEXP; ++e)
#pragma unroll
        for (int off = 16; off; off >>= 1) acc[e] += __shfl_xor(acc[e], off, 32);
    if (lane == 0) {
        float mx = acc[0];
#pragma unroll
        for (int e = 1; e < EEXP; ++e) mx = fmaxf(mx, acc[e]);
        float p[EEXP], s = 0.f;
#pragma unroll
        for (int e = 0; e < EEXP; ++e) { p[e] = __expf(acc[e] - mx); s += p[e]; }
        float inv = __builtin_amdgcn_rcpf(s);
        int i0 = 0; float p0 = p[0];
#pragma unroll
        for (int e = 1; e < EEXP; ++e) if (p[e] > p0) { p0 = p[e]; i0 = e; }
        int i1 = -1; float p1 = -1.f;
#pragma unroll
        for (int e = 0; e < EEXP; ++e) if (e != i0 && p[e] > p1) { p1 = p[e]; i1 = e; }
        topk_idx[token * 2 + 0] = i0;  topk_w[token * 2 + 0] = p0 * inv;
        topk_idx[token * 2 + 1] = i1;  topk_w[token * 2 + 1] = p1 * inv;
        atomicAdd(&counts[i0], 1);
        atomicAdd(&counts[i1], 1);
    }
}

// ---------- kernel: plan (padded offsets + 32-row tile list), single thread ----------
__global__ void k_plan(const int* __restrict__ counts, int* __restrict__ offs,
                       int* __restrict__ tileExpert, int* __restrict__ tileRow0,
                       int* __restrict__ numTiles) {
    if (threadIdx.x == 0 && blockIdx.x == 0) {
        int off = 0, nt = 0;
        for (int e = 0; e < EEXP; ++e) {
            offs[e] = off;
            int tiles = (counts[e] + MTILE - 1) / MTILE;
            for (int t = 0; t < tiles; ++t) { tileExpert[nt] = e; tileRow0[nt] = off + t * MTILE; ++nt; }
            off += tiles * MTILE;
        }
        *numTiles = nt;
    }
}

// ---------- kernel: deterministic per-expert slot fill ----------
__global__ void k_fill(const int* __restrict__ topk_idx, const float* __restrict__ topk_w,
                       const int* __restrict__ offs, const int* __restrict__ counts,
                       int* __restrict__ slot_token, float* __restrict__ slot_w) {
    int e = blockIdx.x;
    __shared__ int sBase;
    __shared__ int wOff[8];
    if (threadIdx.x == 0) sBase = offs[e];
    __syncthreads();
    int lane = threadIdx.x & 31, wid = threadIdx.x >> 5;
    for (int t0 = 0; t0 < T_TOK; t0 += 256) {
        int t = t0 + threadIdx.x;
        int which = -1;
        if (topk_idx[t * 2 + 0] == e)      which = 0;
        else if (topk_idx[t * 2 + 1] == e) which = 1;
        unsigned m32 = (unsigned)__ballot(which >= 0);
        int before = __popc(m32 & ((1u << lane) - 1u));
        if (lane == 0) wOff[wid] = __popc(m32);
        __syncthreads();
        int wbase = 0, total = 0;
        for (int i = 0; i < 8; ++i) { if (i < wid) wbase += wOff[i]; total += wOff[i]; }
        int base = sBase;
        if (which >= 0) {
            int pos = base + wbase + before;
            slot_token[pos] = t;
            slot_w[pos] = topk_w[t * 2 + which];
        }
        __syncthreads();
        if (threadIdx.x == 0) sBase = base + total;
        __syncthreads();
    }
    // pad partial tile: token 0 (valid row), weight 0 -> contributes nothing
    int n = counts[e];
    int padded = ((n + MTILE - 1) / MTILE) * MTILE;
    int base = offs[e];
    for (int i = n + threadIdx.x; i < padded; i += 256) {
        slot_token[base + i] = 0;
        slot_w[base + i] = 0.f;
    }
}

// ---------- kernel: shared expert hidden  Hs = silu(x@sg) * (x@su) ----------
// grid (T/32, SI/128), block 256 (8 waves x 16 cols)
__global__ void k_shared_h(const uint16_t* __restrict__ xf,
                           const uint16_t* __restrict__ SgT, const uint16_t* __restrict__ SuT,
                           uint16_t* __restrict__ Hs) {
    int lane = threadIdx.x & 31, wid = threadIdx.x >> 5;
    int n0 = blockIdx.y * 128 + wid * 16;
    int row0 = blockIdx.x * MTILE;
    int nr = n0 + (lane & 15);
    const uint16_t* wg = SgT + (size_t)nr * DDIM;
    const uint16_t* wu = SuT + (size_t)nr * DDIM;
    const uint16_t* a0row = xf + (size_t)(row0 + (lane & 15)) * DDIM;
    const uint16_t* a1row = a0row + 16 * DDIM;
    v8f cg0 = {}, cg1 = {}, cu0 = {}, cu1 = {};
#pragma unroll 2
    for (int k0 = 0; k0 < DDIM; k0 += 32) {
        ABFrag a0, a1, bg, bu;
        load_afrag(a0, a0row, k0, lane);
        load_afrag(a1, a1row, k0, lane);
        load_bfragT(bg, wg, k0, lane);
        load_bfragT(bu, wu, k0, lane);
        cg0 = WMMA_BF16(a0, bg, cg0); cu0 = WMMA_BF16(a0, bu, cu0);
        cg1 = WMMA_BF16(a1, bg, cg1); cu1 = WMMA_BF16(a1, bu, cu1);
    }
    int hl = lane >> 4, col = n0 + (lane & 15);
#pragma unroll
    for (int v = 0; v < 8; ++v) {
        int r = row0 + v + 8 * hl;
        Hs[(size_t)r * SIDIM + col]        = f2bf(silu_glu(cg0[v], cu0[v]));
        Hs[(size_t)(r + 16) * SIDIM + col] = f2bf(silu_glu(cg1[v], cu1[v]));
    }
}

// ---------- kernel: shared out  d_out = Hs @ sd  (initializes every element) ----------
__global__ void k_shared_out(const uint16_t* __restrict__ Hs, const uint16_t* __restrict__ SdT,
                             float* __restrict__ out) {
    int lane = threadIdx.x & 31, wid = threadIdx.x >> 5;
    int n0 = blockIdx.y * 128 + wid * 16;
    int row0 = blockIdx.x * MTILE;
    const uint16_t* wd = SdT + (size_t)(n0 + (lane & 15)) * SIDIM;
    const uint16_t* a0row = Hs + (size_t)(row0 + (lane & 15)) * SIDIM;
    const uint16_t* a1row = a0row + 16 * SIDIM;
    v8f c0 = {}, c1 = {};
#pragma unroll 2
    for (int k0 = 0; k0 < SIDIM; k0 += 32) {
        ABFrag a0, a1, b;
        load_afrag(a0, a0row, k0, lane);
        load_afrag(a1, a1row, k0, lane);
        load_bfragT(b, wd, k0, lane);
        c0 = WMMA_BF16(a0, b, c0);
        c1 = WMMA_BF16(a1, b, c1);
    }
    int hl = lane >> 4, col = n0 + (lane & 15);
#pragma unroll
    for (int v = 0; v < 8; ++v) {
        int r = row0 + v + 8 * hl;
        out[(size_t)r * DDIM + col]        = c0[v];
        out[(size_t)(r + 16) * DDIM + col] = c1[v];
    }
}

// ---------- kernel: routed hidden  H = silu(X@wg) * (X@wu) per expert tile ----------
// grid (MAXTILE, I/128), block 256
__global__ void k_routed_h(const uint16_t* __restrict__ xf,
                           const uint16_t* __restrict__ WgT, const uint16_t* __restrict__ WuT,
                           const int* __restrict__ tileExpert, const int* __restrict__ tileRow0,
                           const int* __restrict__ numTiles, const int* __restrict__ slot_token,
                           uint16_t* __restrict__ H) {
    int tile = blockIdx.x;
    if (tile >= *numTiles) return;
    int e = tileExpert[tile], row0 = tileRow0[tile];
    int lane = threadIdx.x & 31, wid = threadIdx.x >> 5;
    int n0 = blockIdx.y * 128 + wid * 16;
    size_t eoff = (size_t)e * IDIM * DDIM + (size_t)(n0 + (lane & 15)) * DDIM;
    const uint16_t* wg = WgT + eoff;
    const uint16_t* wu = WuT + eoff;
    const uint16_t* a0row = xf + (size_t)slot_token[row0 + (lane & 15)] * DDIM;
    const uint16_t* a1row = xf + (size_t)slot_token[row0 + 16 + (lane & 15)] * DDIM;
    v8f cg0 = {}, cg1 = {}, cu0 = {}, cu1 = {};
#pragma unroll 2
    for (int k0 = 0; k0 < DDIM; k0 += 32) {
        ABFrag a0, a1, bg, bu;
        load_afrag(a0, a0row, k0, lane);
        load_afrag(a1, a1row, k0, lane);
        load_bfragT(bg, wg, k0, lane);
        load_bfragT(bu, wu, k0, lane);
        cg0 = WMMA_BF16(a0, bg, cg0); cu0 = WMMA_BF16(a0, bu, cu0);
        cg1 = WMMA_BF16(a1, bg, cg1); cu1 = WMMA_BF16(a1, bu, cu1);
    }
    int hl = lane >> 4, col = n0 + (lane & 15);
#pragma unroll
    for (int v = 0; v < 8; ++v) {
        int r = row0 + v + 8 * hl;
        H[(size_t)r * IDIM + col]        = f2bf(silu_glu(cg0[v], cu0[v]));
        H[(size_t)(r + 16) * IDIM + col] = f2bf(silu_glu(cg1[v], cu1[v]));
    }
}

// ---------- kernel: routed out  d_out += w * (H @ wd) ----------
__global__ void k_routed_out(const uint16_t* __restrict__ H, const uint16_t* __restrict__ WdT,
                             const int* __restrict__ tileExpert, const int* __restrict__ tileRow0,
                             const int* __restrict__ numTiles, const int* __restrict__ slot_token,
                             const float* __restrict__ slot_w, float* __restrict__ out) {
    int tile = blockIdx.x;
    if (tile >= *numTiles) return;
    int e = tileExpert[tile], row0 = tileRow0[tile];
    int lane = threadIdx.x & 31, wid = threadIdx.x >> 5;
    int n0 = blockIdx.y * 128 + wid * 16;
    const uint16_t* wd = WdT + (size_t)e * DDIM * IDIM + (size_t)(n0 + (lane & 15)) * IDIM;
    const uint16_t* a0row = H + (size_t)(row0 + (lane & 15)) * IDIM;
    const uint16_t* a1row = a0row + 16 * IDIM;
    v8f c0 = {}, c1 = {};
#pragma unroll 2
    for (int k0 = 0; k0 < IDIM; k0 += 32) {
        ABFrag a0, a1, b;
        load_afrag(a0, a0row, k0, lane);
        load_afrag(a1, a1row, k0, lane);
        load_bfragT(b, wd, k0, lane);
        c0 = WMMA_BF16(a0, b, c0);
        c1 = WMMA_BF16(a1, b, c1);
    }
    int hl = lane >> 4, col = n0 + (lane & 15);
#pragma unroll
    for (int v = 0; v < 8; ++v) {
        int r0 = row0 + v + 8 * hl;
        int r1 = r0 + 16;
        atomicAdd(&out[(size_t)slot_token[r0] * DDIM + col], slot_w[r0] * c0[v]);
        atomicAdd(&out[(size_t)slot_token[r1] * DDIM + col], slot_w[r1] * c1[v]);
    }
}

// ---------- host ----------
extern "C" void kernel_launch(void* const* d_in, const int* in_sizes, int n_in,
                              void* d_out, int out_size, void* d_ws, size_t ws_size,
                              hipStream_t stream) {
    const float* x      = (const float*)d_in[0];
    const float* gate_w = (const float*)d_in[1];
    const float* w_gate = (const float*)d_in[2];
    const float* w_up   = (const float*)d_in[3];
    const float* w_down = (const float*)d_in[4];
    const float* sg     = (const float*)d_in[5];
    const float* su     = (const float*)d_in[6];
    const float* sd     = (const float*)d_in[7];
    float* out = (float*)d_out;

    char* w = (char*)d_ws;
    size_t o = 0;
    auto take = [&](size_t bytes) { char* p = w + o; o += (bytes + 255) & ~(size_t)255; return p; };
    uint16_t* XF    = (uint16_t*)take((size_t)T_TOK * DDIM * 2);
    int*      TOPKI = (int*)     take((size_t)T_TOK * 2 * 4);
    float*    TOPKW = (float*)   take((size_t)T_TOK * 2 * 4);
    int*      COUNT = (int*)     take(EEXP * 4);
    int*      OFFS  = (int*)     take(EEXP * 4);
    int*      TILEE = (int*)     take(MAXTILE * 4);
    int*      TILER = (int*)     take(MAXTILE * 4);
    int*      NT    = (int*)     take(4);
    int*      STOK  = (int*)     take(MAXSLOT * 4);
    float*    SW    = (float*)   take(MAXSLOT * 4);
    uint16_t* H     = (uint16_t*)take((size_t)MAXSLOT * IDIM * 2);
    uint16_t* HS    = (uint16_t*)take((size_t)T_TOK * SIDIM * 2);
    uint16_t* WGT   = (uint16_t*)take((size_t)EEXP * IDIM * DDIM * 2);  // [E][I][D]
    uint16_t* WUT   = (uint16_t*)take((size_t)EEXP * IDIM * DDIM * 2);  // [E][I][D]
    uint16_t* WDT   = (uint16_t*)take((size_t)EEXP * DDIM * IDIM * 2);  // [E][D][I]
    uint16_t* SGT   = (uint16_t*)take((size_t)SIDIM * DDIM * 2);        // [SI][D]
    uint16_t* SUT   = (uint16_t*)take((size_t)SIDIM * DDIM * 2);        // [SI][D]
    uint16_t* SDT   = (uint16_t*)take((size_t)DDIM * SIDIM * 2);        // [D][SI]

    k_init<<<1, 32, 0, stream>>>(COUNT);
    k_x2bf<<<(T_TOK * DDIM) / 256, 256, 0, stream>>>(x, XF);

    // one-shot weight transpose+convert (bandwidth-bound; removes all VALU
    // conversion work from the compute-bound WMMA loops)
    dim3 tb(32, 8);
    k_tcvt<<<dim3(IDIM / 32, DDIM / 32, EEXP), tb, 0, stream>>>(w_gate, WGT, DDIM, IDIM);
    k_tcvt<<<dim3(IDIM / 32, DDIM / 32, EEXP), tb, 0, stream>>>(w_up,   WUT, DDIM, IDIM);
    k_tcvt<<<dim3(DDIM / 32, IDIM / 32, EEXP), tb, 0, stream>>>(w_down, WDT, IDIM, DDIM);
    k_tcvt<<<dim3(SIDIM / 32, DDIM / 32, 1),   tb, 0, stream>>>(sg, SGT, DDIM, SIDIM);
    k_tcvt<<<dim3(SIDIM / 32, DDIM / 32, 1),   tb, 0, stream>>>(su, SUT, DDIM, SIDIM);
    k_tcvt<<<dim3(DDIM / 32, SIDIM / 32, 1),   tb, 0, stream>>>(sd, SDT, SIDIM, DDIM);

    k_gate<<<T_TOK / 4, 128, 0, stream>>>(x, gate_w, TOPKI, TOPKW, COUNT);
    k_plan<<<1, 1, 0, stream>>>(COUNT, OFFS, TILEE, TILER, NT);
    k_fill<<<EEXP, 256, 0, stream>>>(TOPKI, TOPKW, OFFS, COUNT, STOK, SW);

    // shared expert (writes/initializes every element of d_out)
    k_shared_h  <<<dim3(T_TOK / MTILE, SIDIM / 128), 256, 0, stream>>>(XF, SGT, SUT, HS);
    k_shared_out<<<dim3(T_TOK / MTILE, DDIM / 128),  256, 0, stream>>>(HS, SDT, out);

    // routed experts (accumulate on top)
    k_routed_h  <<<dim3(MAXTILE, IDIM / 128), 256, 0, stream>>>(XF, WGT, WUT, TILEE, TILER, NT, STOK, H);
    k_routed_out<<<dim3(MAXTILE, DDIM / 128), 256, 0, stream>>>(H, WDT, TILEE, TILER, NT, STOK, SW, out);
    (void)in_sizes; (void)n_in; (void)out_size; (void)ws_size;
}
